// PointNet2_43782896615733
// MI455X (gfx1250) — matrix operand order repaired and checked by reading, not compiled, not run on hardware
//
#include <hip/hip_runtime.h>
#include <hip/hip_bf16.h>

typedef __attribute__((ext_vector_type(16))) _Float16 v16h;
typedef __attribute__((ext_vector_type(8)))  _Float16 v8h;
typedef __attribute__((ext_vector_type(8)))  float    v8f;

#define EPSBN 1e-5f

// ---------------- Farthest point sampling (one block per batch) ----------------
__global__ void fps_kernel(const float* __restrict__ xyz, int stride, int N, int npoint,
                           int* __restrict__ fidx) {
  __shared__ float dist[4096];
  __shared__ float redv[256];
  __shared__ int   redi[256];
  __shared__ int   sel;
  const int b = blockIdx.x, t = threadIdx.x;
  const float* base = xyz + (size_t)b * N * stride;
  for (int i = t; i < N; i += 256) dist[i] = 1e10f;
  if (t == 0) sel = 0;
  __syncthreads();
  for (int j = 0; j < npoint; ++j) {
    const int c = sel;
    if (t == 0) fidx[b * npoint + j] = c;
    const float cx = base[(size_t)c * stride + 0];
    const float cy = base[(size_t)c * stride + 1];
    const float cz = base[(size_t)c * stride + 2];
    float bv = -1.0f; int bi = 0;
    for (int i = t; i < N; i += 256) {
      const float dx = base[(size_t)i * stride + 0] - cx;
      const float dy = base[(size_t)i * stride + 1] - cy;
      const float dz = base[(size_t)i * stride + 2] - cz;
      const float d = dx * dx + dy * dy + dz * dz;
      const float m = fminf(dist[i], d);
      dist[i] = m;
      if (m > bv) { bv = m; bi = i; }   // keeps smallest index on ties (i ascending)
    }
    redv[t] = bv; redi[t] = bi;
    __syncthreads();
    for (int s = 128; s > 0; s >>= 1) {
      if (t < s) {
        if (redv[t + s] > redv[t] ||
            (redv[t + s] == redv[t] && redi[t + s] < redi[t])) {
          redv[t] = redv[t + s]; redi[t] = redi[t + s];
        }
      }
      __syncthreads();
    }
    if (t == 0) sel = redi[0];
    __syncthreads();
  }
}

// ---------------- Gather sampled centers ----------------
__global__ void gather_xyz_kernel(const float* __restrict__ src, int stride, int N,
                                  const int* __restrict__ fidx, int S,
                                  float* __restrict__ dst, int B) {
  const int id = blockIdx.x * blockDim.x + threadIdx.x;
  if (id >= B * S) return;
  const int b = id / S;
  const int n = fidx[id];
  const float* p = src + ((size_t)b * N + n) * stride;
  dst[(size_t)id * 3 + 0] = p[0];
  dst[(size_t)id * 3 + 1] = p[1];
  dst[(size_t)id * 3 + 2] = p[2];
}

// ---------------- Ball query: first nsample indices within radius, pad with first ----
__global__ void ball_query_kernel(const float* __restrict__ xyz, int stride, int N,
                                  const float* __restrict__ new_xyz, int* __restrict__ idx,
                                  int S, int K, float r2, int B) {
  const int id = blockIdx.x * blockDim.x + threadIdx.x;
  if (id >= B * S) return;
  const int b = id / S;
  const float nx = new_xyz[(size_t)id * 3 + 0];
  const float ny = new_xyz[(size_t)id * 3 + 1];
  const float nz = new_xyz[(size_t)id * 3 + 2];
  const float* base = xyz + (size_t)b * N * stride;
  int cnt = 0, first = 0; bool has = false;
  for (int n = 0; n < N && cnt < K; ++n) {
    const float dx = base[(size_t)n * stride + 0] - nx;
    const float dy = base[(size_t)n * stride + 1] - ny;
    const float dz = base[(size_t)n * stride + 2] - nz;
    if (dx * dx + dy * dy + dz * dz <= r2) {
      if (!has) { first = n; has = true; }
      idx[(size_t)id * K + cnt++] = n;
    }
  }
  for (; cnt < K; ++cnt) idx[(size_t)id * K + cnt] = first;
}

// ---------------- Grouping: build padded f16 GEMM input rows ----------------
__global__ void group_kernel(const float* __restrict__ xyz, int xst,
                             const float* __restrict__ pts, int pst, int poff,
                             const float* __restrict__ new_xyz, const int* __restrict__ idx,
                             _Float16* __restrict__ A, int S, int K, int Nsrc,
                             int Cp, int Kp, int B) {
  const int r = blockIdx.x * blockDim.x + threadIdx.x;
  if (r >= B * S * K) return;
  const int k = r % K;
  const int bs = r / K;
  const int b = bs / S;
  const int n = idx[(size_t)bs * K + k];
  const float* q  = xyz + ((size_t)b * Nsrc + n) * xst;
  const float* nz = new_xyz + (size_t)bs * 3;
  _Float16* a = A + (size_t)r * Kp;
  a[0] = (_Float16)(q[0] - nz[0]);
  a[1] = (_Float16)(q[1] - nz[1]);
  a[2] = (_Float16)(q[2] - nz[2]);
  const float* p = pts + ((size_t)b * Nsrc + n) * pst + poff;
  for (int c = 0; c < Cp; ++c) a[3 + c] = (_Float16)p[c];
  for (int c = 3 + Cp; c < Kp; ++c) a[c] = (_Float16)0.f;
}

// ---------------- SA3 group_all: concat(l2_xyz, l2_pts), pad 259 -> 288 ----------------
__global__ void group_all_kernel(const float* __restrict__ xyz2,
                                 const float* __restrict__ pts,
                                 _Float16* __restrict__ A) {
  const int r = blockIdx.x * blockDim.x + threadIdx.x;
  if (r >= 32 * 128) return;
  _Float16* a = A + (size_t)r * 288;
  a[0] = (_Float16)xyz2[(size_t)r * 3 + 0];
  a[1] = (_Float16)xyz2[(size_t)r * 3 + 1];
  a[2] = (_Float16)xyz2[(size_t)r * 3 + 2];
  const float* p = pts + (size_t)r * 256;
  for (int c = 0; c < 256; ++c) a[3 + c] = (_Float16)p[c];
  for (int c = 259; c < 288; ++c) a[c] = (_Float16)0.f;
}

// ---------------- Weight convert: (Cin x Nout) f32 -> transposed padded (Nout x Kp) f16 ----
__global__ void wt_kernel(const float* __restrict__ W, _Float16* __restrict__ Wt,
                          int Cin, int Nout, int Kp) {
  const int id = blockIdx.x * blockDim.x + threadIdx.x;
  if (id >= Nout * Kp) return;
  const int n = id / Kp;
  const int k = id % Kp;
  Wt[id] = (k < Cin) ? (_Float16)W[(size_t)k * Nout + n] : (_Float16)0.f;
}

// ---------------- WMMA GEMM: H(MxN,f32) = A(MxKp,f16) * Wt(NxKp,f16)^T ----------------
// Each wave computes a 16x64 output strip (4 accumulators): the A fragment is loaded
// once per 32-wide K chunk and reused by 4 v_wmma_f32_16x16x32_f16 issues.
__global__ void gemm_wmma_kernel(const _Float16* __restrict__ A,
                                 const _Float16* __restrict__ Wt,
                                 float* __restrict__ H, int M, int N, int Kp) {
  const int wave = threadIdx.x >> 5;
  const int lane = threadIdx.x & 31;
  const int tilesM  = M >> 4;
  const int tilesN4 = N >> 6;                 // N is always a multiple of 64 here
  const int tile = blockIdx.x * 8 + wave;
  if (tile >= tilesM * tilesN4) return;       // wave-uniform: EXEC all-1s inside
  const int tm = tile % tilesM;
  const int tn = (tile / tilesM) << 2;        // first of 4 adjacent 16-wide N tiles
  const int row = lane & 15;
  const int hi  = lane >> 4;
  // A fragment lane base: 16-bit A 16x32 layout (lanes 0-15: K 0-7/16-23; lanes 16-31: +8)
  const _Float16* arow  = A  + (size_t)(tm * 16 + row) * Kp + hi * 8;
  // B fragment lane base: 16-bit B 32x16 layout (lanes 0-15: K 0-7/16-23; lanes 16-31: +8)
  const _Float16* brow0 = Wt + (size_t)(tn * 16 + row) * Kp + hi * 8;
  v8f acc[4] = {v8f{}, v8f{}, v8f{}, v8f{}};
  for (int kc = 0; kc < Kp; kc += 32) {
    if (kc + 32 < Kp) {                       // WGP-scope prefetch of the next K-chunk
      __builtin_prefetch(arow + kc + 32, 0, 3);
      __builtin_prefetch(brow0 + kc + 32, 0, 3);
    }
    const v8h al = *(const v8h*)(arow + kc);
    const v8h ah = *(const v8h*)(arow + kc + 16);
    v16h a;
#pragma unroll
    for (int i = 0; i < 8; ++i) { a[i] = al[i]; a[i + 8] = ah[i]; }
#pragma unroll
    for (int j = 0; j < 4; ++j) {
      const _Float16* brow = brow0 + (size_t)j * 16 * Kp;
      const v8h b0 = *(const v8h*)(brow + kc);
      const v8h b1 = *(const v8h*)(brow + kc + 16);
      v16h bb;
#pragma unroll
      for (int i = 0; i < 8; ++i) { bb[i] = b0[i]; bb[i + 8] = b1[i]; }
      acc[j] = __builtin_amdgcn_wmma_f32_16x16x32_f16(false, a, false, bb, (short)0,
                                                      acc[j], false, false);
    }
  }
  // C/D layout: VGPR r -> M = r + 8*hi, col = lane&15
#pragma unroll
  for (int j = 0; j < 4; ++j) {
    float* hout = H + (size_t)(tm * 16 + 8 * hi) * N + (tn + j) * 16 + row;
#pragma unroll
    for (int r = 0; r < 8; ++r) hout[(size_t)r * N] = acc[j][r];
  }
}

// ---------------- BN stats: one block per channel ----------------
__global__ void bn_stats_kernel(const float* __restrict__ H, int M, int N,
                                float* __restrict__ mean, float* __restrict__ var) {
  __shared__ float s1[256];
  __shared__ float s2[256];
  const int n = blockIdx.x, t = threadIdx.x;
  float a = 0.f, b = 0.f;
  for (int r = t; r < M; r += 256) {
    const float v = H[(size_t)r * N + n];
    a += v; b += v * v;
  }
  s1[t] = a; s2[t] = b;
  __syncthreads();
  for (int s = 128; s > 0; s >>= 1) {
    if (t < s) { s1[t] += s1[t + s]; s2[t] += s2[t + s]; }
    __syncthreads();
  }
  if (t == 0) {
    const float m = s1[0] / (float)M;
    mean[n] = m;
    var[n] = s2[0] / (float)M - m * m;
  }
}

// ---------------- BN normalize + ReLU, in-place f32 + f16 recast for next GEMM ----------
__global__ void bn_norm_kernel(float* __restrict__ H, _Float16* __restrict__ X,
                               const float* __restrict__ g, const float* __restrict__ bt,
                               const float* __restrict__ mean, const float* __restrict__ var,
                               int M, int N) {
  const size_t id = (size_t)blockIdx.x * blockDim.x + threadIdx.x;
  if (id >= (size_t)M * N) return;
  const int n = (int)(id % (size_t)N);
  float y = g[n] * (H[id] - mean[n]) * rsqrtf(var[n] + EPSBN) + bt[n];
  y = fmaxf(y, 0.f);
  H[id] = y;
  X[id] = (_Float16)y;
}

// ---------------- Max-pool over the nsample axis ----------------
__global__ void maxpool_kernel(const float* __restrict__ H, float* __restrict__ out,
                               int BS, int K, int C) {
  const int id = blockIdx.x * blockDim.x + threadIdx.x;
  if (id >= BS * C) return;
  const int bs = id / C;
  const int c = id % C;
  const float* p = H + (size_t)bs * K * C + c;
  float m = p[0];
  for (int k = 1; k < K; ++k) m = fmaxf(m, p[(size_t)k * C]);
  out[id] = m;
}

__global__ void cast_kernel(const float* __restrict__ src, _Float16* __restrict__ dst, int n) {
  const int id = blockIdx.x * blockDim.x + threadIdx.x;
  if (id < n) dst[id] = (_Float16)src[id];
}

__global__ void copy_kernel(const float* __restrict__ src, float* __restrict__ dst, int n) {
  const int id = blockIdx.x * blockDim.x + threadIdx.x;
  if (id < n) dst[id] = src[id];
}

// ---------------- Final tiny FC: out = feat @ w3 + b3  (32x64 x 64x2) ----------------
__global__ void fc_out_kernel(const float* __restrict__ feat, const float* __restrict__ w3,
                              const float* __restrict__ b3, float* __restrict__ out) {
  const int id = threadIdx.x;
  if (id >= 64) return;
  const int b = id >> 1, j = id & 1;
  float s = b3[j];
  for (int c = 0; c < 64; ++c) s += feat[b * 64 + c] * w3[c * 2 + j];
  out[b * 2 + j] = s;
}

extern "C" void kernel_launch(void* const* d_in, const int* in_sizes, int n_in,
                              void* d_out, int out_size, void* d_ws, size_t ws_size,
                              hipStream_t stream) {
  (void)in_sizes; (void)n_in; (void)out_size; (void)ws_size;
  const float* xyz = (const float*)d_in[0];
  auto W = [&](int i) { return (const float*)d_in[i]; };

  // ---- workspace carve-out ----
  char* ws = (char*)d_ws;
  size_t off = 0;
  auto alloc = [&](size_t bytes) -> void* {
    off = (off + 255) & ~(size_t)255;
    void* p = ws + off;
    off += bytes;
    return p;
  };
  const int B = 32, N = 4096;
  int*      fidx1  = (int*)alloc((size_t)B * 512 * 4);
  float*    nxyz1  = (float*)alloc((size_t)B * 512 * 3 * 4);
  int*      bidx1  = (int*)alloc((size_t)B * 512 * 32 * 4);
  int*      fidx2  = (int*)alloc((size_t)B * 128 * 4);
  float*    nxyz2  = (float*)alloc((size_t)B * 128 * 3 * 4);
  int*      bidx2  = (int*)alloc((size_t)B * 128 * 64 * 4);
  _Float16* Abuf   = (_Float16*)alloc((size_t)262144 * 160 * 2);   // max padded GEMM input
  _Float16* Wt     = (_Float16*)alloc((size_t)512 * 1024 * 2);     // max transposed weight
  float*    H      = (float*)alloc((size_t)67108864 * 4);          // max M*N activations
  _Float16* X      = (_Float16*)alloc((size_t)67108864 * 2);       // f16 next-layer input
  float*    pooled1 = (float*)alloc((size_t)B * 512 * 128 * 4);
  float*    pooled2 = (float*)alloc((size_t)B * 128 * 256 * 4);
  float*    l3      = (float*)alloc((size_t)B * 1024 * 4);
  float*    mean    = (float*)alloc(1024 * 4);
  float*    var     = (float*)alloc(1024 * 4);

  auto cdiv = [](long a, long b) { return (int)((a + b - 1) / b); };

  // conv/linear bias is cancelled by train-mode BN mean subtraction -> skip it
  auto run_layer = [&](const _Float16* Ain, int M, int Kp, int Cin,
                       const float* w, const float* g, const float* bt, int Nout) {
    wt_kernel<<<cdiv((long)Nout * Kp, 256), 256, 0, stream>>>(w, Wt, Cin, Nout, Kp);
    const int tiles = (M / 16) * (Nout / 64);   // one 16x64 strip per wave
    gemm_wmma_kernel<<<cdiv(tiles, 8), 256, 0, stream>>>(Ain, Wt, H, M, Nout, Kp);
    bn_stats_kernel<<<Nout, 256, 0, stream>>>(H, M, Nout, mean, var);
    bn_norm_kernel<<<cdiv((long)M * Nout, 256), 256, 0, stream>>>(H, X, g, bt, mean, var, M, Nout);
  };

  // ================= SA1: npoint=512, r=0.2, K=32, 6 -> 64 -> 64 -> 128 ===========
  fps_kernel<<<B, 256, 0, stream>>>(xyz, 6, N, 512, fidx1);
  gather_xyz_kernel<<<cdiv(B * 512, 256), 256, 0, stream>>>(xyz, 6, N, fidx1, 512, nxyz1, B);
  ball_query_kernel<<<cdiv(B * 512, 256), 256, 0, stream>>>(xyz, 6, N, nxyz1, bidx1, 512, 32, 0.04f, B);
  group_kernel<<<cdiv((long)B * 512 * 32, 256), 256, 0, stream>>>(
      xyz, 6, xyz, 6, 3, nxyz1, bidx1, Abuf, 512, 32, N, 3, 32, B);
  run_layer(Abuf, B * 512 * 32, 32, 6,   W(1), W(3),  W(4),  64);
  run_layer(X,    B * 512 * 32, 64, 64,  W(5), W(7),  W(8),  64);
  run_layer(X,    B * 512 * 32, 64, 64,  W(9), W(11), W(12), 128);
  maxpool_kernel<<<cdiv((long)B * 512 * 128, 256), 256, 0, stream>>>(H, pooled1, B * 512, 32, 128);

  // ================= SA2: npoint=128, r=0.4, K=64, 131 -> 128 -> 128 -> 256 =======
  fps_kernel<<<B, 256, 0, stream>>>(nxyz1, 3, 512, 128, fidx2);
  gather_xyz_kernel<<<cdiv(B * 128, 256), 256, 0, stream>>>(nxyz1, 3, 512, fidx2, 128, nxyz2, B);
  ball_query_kernel<<<cdiv(B * 128, 256), 256, 0, stream>>>(nxyz1, 3, 512, nxyz2, bidx2, 128, 64, 0.16f, B);
  group_kernel<<<cdiv((long)B * 128 * 64, 256), 256, 0, stream>>>(
      nxyz1, 3, pooled1, 128, 0, nxyz2, bidx2, Abuf, 128, 64, 512, 128, 160, B);
  run_layer(Abuf, B * 128 * 64, 160, 131, W(13), W(15), W(16), 128);
  run_layer(X,    B * 128 * 64, 128, 128, W(17), W(19), W(20), 128);
  run_layer(X,    B * 128 * 64, 128, 128, W(21), W(23), W(24), 256);
  maxpool_kernel<<<cdiv((long)B * 128 * 256, 256), 256, 0, stream>>>(H, pooled2, B * 128, 64, 256);

  // ================= SA3: group_all, 259 -> 256 -> 512 -> 1024 ====================
  group_all_kernel<<<cdiv(B * 128, 256), 256, 0, stream>>>(nxyz2, pooled2, Abuf);
  run_layer(Abuf, B * 128, 288, 259, W(25), W(27), W(28), 256);
  run_layer(X,    B * 128, 256, 256, W(29), W(31), W(32), 512);
  run_layer(X,    B * 128, 512, 512, W(33), W(35), W(36), 1024);
  maxpool_kernel<<<cdiv((long)B * 1024, 256), 256, 0, stream>>>(H, l3, B, 128, 1024);

  float* out = (float*)d_out;
  // l3_pts -> d_out[2112:]
  copy_kernel<<<cdiv(B * 1024, 256), 256, 0, stream>>>(l3, out + 64 + B * 64, B * 1024);

  // ================= FC head: 1024 -> 256 -> 64 -> 2 ==============================
  cast_kernel<<<cdiv(B * 1024, 256), 256, 0, stream>>>(l3, Abuf, B * 1024);
  run_layer(Abuf, B, 1024, 1024, W(37), W(39), W(40), 256);
  run_layer(X,    B, 256,  256,  W(41), W(43), W(44), 64);
  // feat -> d_out[64:2112]
  copy_kernel<<<cdiv(B * 64, 256), 256, 0, stream>>>(H, out + 64, B * 64);
  // out -> d_out[0:64]
  fc_out_kernel<<<1, 64, 0, stream>>>(H, W(45), W(46), out);
}
